// HDCProcessor_65257733095812
// MI455X (gfx1250) — matrix operand order repaired and checked by reading, not compiled, not run on hardware
//
#include <hip/hip_runtime.h>
#include <hip/hip_bf16.h>
#include <stdint.h>

#define D_DIM 4096
#define S_DIM 128
#define B_DIM 4
#define T_DIM 2048
#define M_ROWS (B_DIM * T_DIM)   // 8192

typedef __bf16 bf16_t;
typedef bf16_t bf16x16 __attribute__((ext_vector_type(16)));
typedef bf16_t bf16x8  __attribute__((ext_vector_type(8)));
typedef float  f32x8   __attribute__((ext_vector_type(8)));
typedef float  f32x4   __attribute__((ext_vector_type(4)));

union BF16x16U { bf16x16 v; bf16x8 h[2]; };

// ---------------------------------------------------------------------------
// 1) x = 0.7 * chars*roll(m1,1)*roll(m2,2) + 0.3 * chars   (f32, B*T x D)
// ---------------------------------------------------------------------------
__global__ void k_trigram(const int* __restrict__ idx, const float* __restrict__ cb,
                          float* __restrict__ x) {
  int bt = blockIdx.x;                 // 0..8191 = b*T + t
  int t  = bt & (T_DIM - 1);
  int i0 = idx[bt];
  int i1 = (t >= 1) ? idx[bt - 1] : 0;
  int i2 = (t >= 2) ? idx[bt - 2] : 0;
  float s1 = (t >= 1) ? 1.0f : 0.0f;
  float s2 = (t >= 2) ? 1.0f : 0.0f;
  const float* r0 = cb + (size_t)i0 * D_DIM;
  const float* r1 = cb + (size_t)i1 * D_DIM;
  const float* r2 = cb + (size_t)i2 * D_DIM;
  float* xrow = x + (size_t)bt * D_DIM;
  for (int d = threadIdx.x; d < D_DIM; d += blockDim.x) {
    float c0 = r0[d];
    float c1 = s1 * r1[(d + D_DIM - 1) & (D_DIM - 1)];
    float c2 = s2 * r2[(d + D_DIM - 2) & (D_DIM - 1)];
    xrow[d] = 0.7f * (c0 * c1 * c2) + 0.3f * c0;
  }
}

// ---------------------------------------------------------------------------
// 2) Weight prep: Wt[w][n][k] = bf16(W_w[k][n]) for Wq/Wk/Wv/Wg (n<128,k<4096)
//    Wot[n][k]   = bf16(Wo[k][n])               (n<4096,k<128)
// ---------------------------------------------------------------------------
__global__ void k_prep_w(const float* __restrict__ Wq, const float* __restrict__ Wk,
                         const float* __restrict__ Wv, const float* __restrict__ Wg,
                         const float* __restrict__ Wo,
                         bf16_t* __restrict__ Wt, bf16_t* __restrict__ Wot) {
  int i = blockIdx.x * blockDim.x + threadIdx.x;
  if (i < 4 * S_DIM * D_DIM) {
    int w   = i >> 19;                       // / (128*4096)
    int rem = i & (S_DIM * D_DIM - 1);
    int n   = rem >> 12;                     // / 4096
    int kk  = rem & (D_DIM - 1);
    const float* W = (w == 0) ? Wq : (w == 1) ? Wk : (w == 2) ? Wv : Wg;
    Wt[(size_t)i] = (bf16_t)W[(size_t)kk * S_DIM + n];
  }
  if (i < D_DIM * S_DIM) {
    int n = i >> 7, kk = i & (S_DIM - 1);
    Wot[i] = (bf16_t)Wo[(size_t)kk * D_DIM + n];
  }
}

// ---------------------------------------------------------------------------
// 3) Projections [q|k|v|g] = x @ [Wq|Wk|Wv|Wg] + b (sigmoid on g).
//    WMMA bf16 16x16x32 against the concatenated N=512.
//    grid.x = m_tiles (512). 8 waves/block; each wave owns 4 n-tiles and
//    reuses one A fragment for 4 WMMAs per K-step -> x read once from HBM.
// ---------------------------------------------------------------------------
__global__ void __launch_bounds__(256) k_proj(
    const float* __restrict__ x, const bf16_t* __restrict__ Wt,
    const float* __restrict__ bq, const float* __restrict__ bk,
    const float* __restrict__ bv, const float* __restrict__ bg,
    float* __restrict__ q, float* __restrict__ k,
    float* __restrict__ v, float* __restrict__ g) {
  int mt   = blockIdx.x;
  int wave = threadIdx.x >> 5;     // 0..7: owns n-tiles wave*4 .. wave*4+3
  int lane = threadIdx.x & 31;
  int half = lane >> 4;
  int lm   = lane & 15;
  int m = mt * 16 + lm;            // A row for this lane
  const float* arow = x + (size_t)m * D_DIM;

  // 4 B-row pointers (transposed weights, concatenated rows 0..511)
  const bf16_t* brow[4];
#pragma unroll
  for (int j = 0; j < 4; ++j)
    brow[j] = Wt + ((size_t)(wave * 4 + j) * 16 + lm) * D_DIM;

  f32x8 acc[4] = {};
  for (int k0 = 0; k0 < D_DIM; k0 += 32) {
    // A fragment: K = k0 + 16*half + e (contiguous), f32 -> bf16
    f32x4 a0 = *(const f32x4*)(arow + k0 + half * 16);
    f32x4 a1 = *(const f32x4*)(arow + k0 + half * 16 + 4);
    f32x4 a2 = *(const f32x4*)(arow + k0 + half * 16 + 8);
    f32x4 a3 = *(const f32x4*)(arow + k0 + half * 16 + 12);
    if (k0 + 128 < D_DIM) __builtin_prefetch(arow + k0 + 128, 0, 0);
    bf16x16 a;
    a[0]  = (bf16_t)a0[0];  a[1]  = (bf16_t)a0[1];  a[2]  = (bf16_t)a0[2];  a[3]  = (bf16_t)a0[3];
    a[4]  = (bf16_t)a1[0];  a[5]  = (bf16_t)a1[1];  a[6]  = (bf16_t)a1[2];  a[7]  = (bf16_t)a1[3];
    a[8]  = (bf16_t)a2[0];  a[9]  = (bf16_t)a2[1];  a[10] = (bf16_t)a2[2];  a[11] = (bf16_t)a2[3];
    a[12] = (bf16_t)a3[0];  a[13] = (bf16_t)a3[1];  a[14] = (bf16_t)a3[2];  a[15] = (bf16_t)a3[3];
#pragma unroll
    for (int j = 0; j < 4; ++j) {
      // B fragment: K = k0 + 16*half + e, contiguous in transposed weights
      bf16x16 b = *(const bf16x16*)(brow[j] + k0 + half * 16);
      acc[j] = __builtin_amdgcn_wmma_f32_16x16x32_bf16(false, a, false, b,
                                                       (short)0, acc[j],
                                                       false, false);
    }
  }

  // w is wave-uniform: tiles wave*4..wave*4+3 all lie in matrix wave>>1
  int w = wave >> 1;
  const float* bias = (w == 0) ? bq : (w == 1) ? bk : (w == 2) ? bv : bg;
  float* outp = (w == 0) ? q : (w == 1) ? k : (w == 2) ? v : g;
#pragma unroll
  for (int j = 0; j < 4; ++j) {
    int n = (((wave & 1) * 4 + j) * 16 + lm);   // column within this matrix
    float bn = bias[n];
#pragma unroll
    for (int r = 0; r < 8; ++r) {
      int mr = mt * 16 + r + half * 8;          // C/D row layout
      float val = acc[j][r] + bn;
      if (w == 3) {
        float e = __expf(-val);
        val = __builtin_amdgcn_rcpf(1.0f + e);  // fast sigmoid
      }
      outp[(size_t)mr * S_DIM + n] = val;
    }
  }
}

// ---------------------------------------------------------------------------
// 4) Gated scan. State columns are independent: block = (batch, 32-col chunk),
//    128 threads = 4 lane-groups x 32 cols; each thread holds 32 state rows.
// ---------------------------------------------------------------------------
__global__ void __launch_bounds__(128) k_scan(
    const float* __restrict__ q, const float* __restrict__ k,
    const float* __restrict__ v, const float* __restrict__ g,
    bf16_t* __restrict__ o) {
  __shared__ float sQ[S_DIM], sK[S_DIM], sG[S_DIM], sP[S_DIM];
  int b       = blockIdx.x >> 2;
  int colbase = (blockIdx.x & 3) * 32;
  int tid = threadIdx.x;
  int dl  = tid & 31;
  int r   = tid >> 5;                 // uniform per wave -> LDS broadcast reads
  int d   = colbase + dl;

  float st[32];
#pragma unroll
  for (int i = 0; i < 32; ++i) st[i] = 0.0f;

  for (int t = 0; t < T_DIM; ++t) {
    size_t row = (size_t)(b * T_DIM + t) * S_DIM;
    sQ[tid] = q[row + tid];
    sK[tid] = k[row + tid];
    sG[tid] = g[row + tid];
    float vd = v[row + d];
    __syncthreads();
    float acc = 0.0f;
#pragma unroll
    for (int i = 0; i < 32; ++i) {
      int s = r * 32 + i;
      st[i] = fmaf(sG[s], st[i], sK[s] * vd);
      acc   = fmaf(sQ[s], st[i], acc);
    }
    sP[tid] = acc;
    __syncthreads();
    if (r == 0) {
      float sum = sP[dl] + sP[32 + dl] + sP[64 + dl] + sP[96 + dl];
      o[row + d] = (bf16_t)sum;
    }
  }
}

// ---------------------------------------------------------------------------
// 5) res = x + o @ Wo + bo.  WMMA bf16, K=128.
//    grid.x = m_tiles(512) * 8 n-supertiles; each wave owns 4 n-tiles,
//    reusing one A fragment for 4 WMMAs per K-step (16 WMMAs/wave).
// ---------------------------------------------------------------------------
__global__ void __launch_bounds__(256) k_outproj(
    const bf16_t* __restrict__ o, const bf16_t* __restrict__ Wot,
    const float* __restrict__ x, const float* __restrict__ bo,
    float* __restrict__ out) {
  int nsup = blockIdx.x & 7;       // 8 super-tiles of 512 columns
  int mt   = blockIdx.x >> 3;
  int wave = threadIdx.x >> 5;
  int lane = threadIdx.x & 31;
  int half = lane >> 4;
  int lm   = lane & 15;
  int ntbase = nsup * 32 + wave * 4;           // first of 4 n-tiles
  int m = mt * 16 + lm;
  const bf16_t* arow = o + (size_t)m * S_DIM;

  const bf16_t* brow[4];
#pragma unroll
  for (int j = 0; j < 4; ++j)
    brow[j] = Wot + (size_t)((ntbase + j) * 16 + lm) * S_DIM;

  f32x8 acc[4] = {};
#pragma unroll
  for (int k0 = 0; k0 < S_DIM; k0 += 32) {
    BF16x16U a;
    a.h[0] = *(const bf16x8*)(arow + k0 + half * 16);
    a.h[1] = *(const bf16x8*)(arow + k0 + half * 16 + 8);
#pragma unroll
    for (int j = 0; j < 4; ++j) {
      bf16x16 bfrag = *(const bf16x16*)(brow[j] + k0 + half * 16);
      acc[j] = __builtin_amdgcn_wmma_f32_16x16x32_bf16(false, a.v, false, bfrag,
                                                       (short)0, acc[j],
                                                       false, false);
    }
  }
#pragma unroll
  for (int j = 0; j < 4; ++j) {
    int n = (ntbase + j) * 16 + lm;
    float bn = bo[n];
#pragma unroll
    for (int r = 0; r < 8; ++r) {
      int mr = mt * 16 + r + half * 8;
      out[(size_t)mr * D_DIM + n] = acc[j][r] + x[(size_t)mr * D_DIM + n] + bn;
    }
  }
}

// ---------------------------------------------------------------------------
// 6) In-place LayerNorm over D per row.
// ---------------------------------------------------------------------------
__global__ void __launch_bounds__(256) k_lnorm(float* __restrict__ out,
                                               const float* __restrict__ gamma,
                                               const float* __restrict__ beta) {
  __shared__ float rs[256], rq[256];
  int row = blockIdx.x;
  float* p = out + (size_t)row * D_DIM;
  int base = threadIdx.x * 16;
  float vals[16];
  float sum = 0.0f, sq = 0.0f;
#pragma unroll
  for (int i = 0; i < 16; i += 4) {
    f32x4 tv = *(const f32x4*)(p + base + i);
#pragma unroll
    for (int j = 0; j < 4; ++j) {
      float xv = tv[j];
      vals[i + j] = xv;
      sum += xv;
      sq  = fmaf(xv, xv, sq);
    }
  }
  rs[threadIdx.x] = sum;
  rq[threadIdx.x] = sq;
  __syncthreads();
  for (int s = 128; s > 0; s >>= 1) {
    if (threadIdx.x < s) {
      rs[threadIdx.x] += rs[threadIdx.x + s];
      rq[threadIdx.x] += rq[threadIdx.x + s];
    }
    __syncthreads();
  }
  float mean = rs[0] * (1.0f / (float)D_DIM);
  float var  = rq[0] * (1.0f / (float)D_DIM) - mean * mean;
  float inv  = rsqrtf(var + 1e-5f);
#pragma unroll
  for (int i = 0; i < 16; ++i) {
    int d = base + i;
    p[d] = (vals[i] - mean) * inv * gamma[d] + beta[d];
  }
}

// ---------------------------------------------------------------------------
extern "C" void kernel_launch(void* const* d_in, const int* in_sizes, int n_in,
                              void* d_out, int out_size, void* d_ws, size_t ws_size,
                              hipStream_t stream) {
  const int*   idx  = (const int*)  d_in[0];
  const float* cb   = (const float*)d_in[1];
  const float* Wq   = (const float*)d_in[2];
  const float* bq   = (const float*)d_in[3];
  const float* Wk   = (const float*)d_in[4];
  const float* bk   = (const float*)d_in[5];
  const float* Wv   = (const float*)d_in[6];
  const float* bv   = (const float*)d_in[7];
  const float* Wg   = (const float*)d_in[8];
  const float* bg   = (const float*)d_in[9];
  const float* Wo   = (const float*)d_in[10];
  const float* bo   = (const float*)d_in[11];
  const float* gamma= (const float*)d_in[12];
  const float* beta = (const float*)d_in[13];
  float* out = (float*)d_out;

  char* ws = (char*)d_ws;
  size_t off = 0;
  float* x = (float*)(ws + off);   off += (size_t)M_ROWS * D_DIM * sizeof(float);   // 134 MB
  float* q = (float*)(ws + off);   off += (size_t)M_ROWS * S_DIM * sizeof(float);
  float* k = (float*)(ws + off);   off += (size_t)M_ROWS * S_DIM * sizeof(float);
  float* v = (float*)(ws + off);   off += (size_t)M_ROWS * S_DIM * sizeof(float);
  float* g = (float*)(ws + off);   off += (size_t)M_ROWS * S_DIM * sizeof(float);
  bf16_t* obuf = (bf16_t*)(ws + off); off += (size_t)M_ROWS * S_DIM * sizeof(bf16_t);
  bf16_t* Wt   = (bf16_t*)(ws + off); off += (size_t)4 * S_DIM * D_DIM * sizeof(bf16_t);
  bf16_t* Wot  = (bf16_t*)(ws + off); off += (size_t)D_DIM * S_DIM * sizeof(bf16_t);
  (void)ws_size; (void)in_sizes; (void)n_in; (void)out_size;

  // 1) trigram features
  k_trigram<<<M_ROWS, 256, 0, stream>>>(idx, cb, x);
  // 2) weight transpose + bf16 conversion
  k_prep_w<<<(4 * S_DIM * D_DIM + 255) / 256, 256, 0, stream>>>(Wq, Wk, Wv, Wg, Wo, Wt, Wot);
  // 3) q/k/v/g projections (WMMA bf16, x streamed once)
  k_proj<<<M_ROWS / 16, 256, 0, stream>>>(x, Wt, bq, bk, bv, bg, q, k, v, g);
  // 4) gated linear-attention scan
  k_scan<<<B_DIM * 4, 128, 0, stream>>>(q, k, v, g, obuf);
  // 5) output projection + residual (WMMA bf16)
  k_outproj<<<(M_ROWS / 16) * 8, 256, 0, stream>>>(obuf, Wot, x, bo, out);
  // 6) layernorm in place on d_out
  k_lnorm<<<M_ROWS, 256, 0, stream>>>(out, gamma, beta);
}
